// MultiHeadAttention_7138235646071
// MI455X (gfx1250) — matrix operand order repaired and checked by reading, not compiled, not run on hardware
//
#include <hip/hip_runtime.h>

typedef __bf16 bf16;
typedef bf16  bf16x8 __attribute__((ext_vector_type(8)));
typedef bf16  v16bf  __attribute__((ext_vector_type(16)));
typedef float v8f    __attribute__((ext_vector_type(8)));

#define D_MODEL   1024
#define N_HEAD    16
#define HEAD_DIM  64
#define BATCH     4
#define SEQ       2048
#define NROWS     (BATCH * SEQ)      // 8192

__device__ __forceinline__ v16bf make16(bf16x8 lo, bf16x8 hi) {
  v16bf r;
#pragma unroll
  for (int i = 0; i < 8; i++) { r[i] = lo[i]; r[i + 8] = hi[i]; }
  return r;
}

__device__ __forceinline__ v8f wmma_bf16(v16bf a, v16bf b, v8f c) {
  return __builtin_amdgcn_wmma_f32_16x16x32_bf16(false, a, false, b, (short)0, c,
                                                 false, false);
}

// ---- CDNA5 async global->LDS DMA (ASYNCcnt-tracked), per-lane 16B copies.
// lds_off = byte offset from LDS_BASE (extern __shared__ arena starts at 0).
__device__ __forceinline__ void async_b128(unsigned lds_off, const bf16* g) {
  asm volatile("global_load_async_to_lds_b128 %0, %1, off"
               :: "v"(lds_off), "v"(g) : "memory");
}
__device__ __forceinline__ void wait_async0() {
  asm volatile("s_wait_asynccnt 0" ::: "memory");
}

// ---------------------------------------------------------------- converts
__global__ void __launch_bounds__(256)
cvt_f32_bf16(const float* __restrict__ s, bf16* __restrict__ d, int n) {
  int i = blockIdx.x * blockDim.x + threadIdx.x;
  if (i < n) d[i] = (bf16)s[i];
}

// W[K][N] f32 -> WT[N][K] bf16  (one-time; GEMM B-tiles then stage with b128 DMA)
__global__ void __launch_bounds__(256)
cvt_transpose(const float* __restrict__ W, bf16* __restrict__ WT, int K, int N) {
  __shared__ float tile[32][33];
  int k0 = blockIdx.x * 32, n0 = blockIdx.y * 32;
  int tx = threadIdx.x, ty = threadIdx.y;          // 32 x 8
#pragma unroll
  for (int i = 0; i < 4; i++)
    tile[ty + i * 8][tx] = W[(size_t)(k0 + ty + i * 8) * N + n0 + tx];
  __syncthreads();
#pragma unroll
  for (int i = 0; i < 4; i++)
    WT[(size_t)(n0 + ty + i * 8) * K + k0 + tx] = (bf16)tile[tx][ty + i * 8];
}

// ---------------------------------------------------------------- GEMM core
// macro tile 64(M) x 128(N), K-step 64, double-buffered async LDS staging.
// LDS arena (bytes): Abuf[2] @ 0 / 10240 (64 rows x 80 elems),
//                    Bbuf[2] @ 20480 / 40960 (128 rows x 80 elems). Total 60 KiB.
#define GEMM_LDS_BYTES 61440
#define A_BUF(b)  ((b) * 10240)
#define B_BUF(b)  (20480 + (b) * 20480)

__device__ __forceinline__ void gemm_stage(const char* /*unused*/, int buf,
                                           const bf16* A, const bf16* WT,
                                           int K, int m0, int n0, int k0, int tid) {
  int ar = tid >> 2, ac = (tid & 3) * 16;          // A: 16 elems/thread
  const bf16* ga = A + (size_t)(m0 + ar) * K + k0 + ac;
  unsigned la = A_BUF(buf) + (unsigned)(ar * 80 + ac) * 2;
  async_b128(la, ga);
  async_b128(la + 16, ga + 8);
  int br = tid >> 1, bc = (tid & 1) * 32;          // B: 32 elems/thread
  const bf16* gb = WT + (size_t)(n0 + br) * K + k0 + bc;
  unsigned lb = B_BUF(buf) + (unsigned)(br * 80 + bc) * 2;
#pragma unroll
  for (int i = 0; i < 4; i++) async_b128(lb + i * 16, gb + i * 8);
}

__device__ __forceinline__ void gemm_compute(const char* smem, int buf,
                                             int wr, int wc, int half, int l15,
                                             v8f acc[2][2]) {
  const char* As = smem + A_BUF(buf);
  const char* Bs = smem + B_BUF(buf);
#pragma unroll
  for (int kc = 0; kc < 64; kc += 32) {
    v16bf af[2], bfr[2];
#pragma unroll
    for (int i = 0; i < 2; i++) {
      int row = wr * 32 + i * 16 + l15;
      af[i] = make16(*(const bf16x8*)(As + (row * 80 + kc + half * 16) * 2),
                     *(const bf16x8*)(As + (row * 80 + kc + half * 16 + 8) * 2));
    }
#pragma unroll
    for (int j = 0; j < 2; j++) {
      int col = wc * 32 + j * 16 + l15;
      bfr[j] = make16(*(const bf16x8*)(Bs + (col * 80 + kc + half * 16) * 2),
                      *(const bf16x8*)(Bs + (col * 80 + kc + half * 16 + 8) * 2));
    }
#pragma unroll
    for (int i = 0; i < 2; i++)
#pragma unroll
      for (int j = 0; j < 2; j++)
        acc[i][j] = wmma_bf16(af[i], bfr[j], acc[i][j]);
  }
}

// ---------------------------------------------------------------- QKV GEMM
__global__ void __launch_bounds__(256)
gemm_qkv(const bf16* __restrict__ A, const bf16* __restrict__ WT,
         const float* __restrict__ bias,
         bf16* __restrict__ Qo, bf16* __restrict__ Ko, bf16* __restrict__ Vt) {
  extern __shared__ char smem[];
  const int K = D_MODEL;
  const int tid  = threadIdx.x;
  const int lane = tid & 31, wave = tid >> 5;
  const int half = lane >> 4, l15 = lane & 15;
  const int wr = wave >> 2, wc = wave & 3;
  const int m0 = blockIdx.x * 64, n0 = blockIdx.y * 128;

  v8f acc[2][2] = {{{}, {}}, {{}, {}}};

  gemm_stage(smem, 0, A, WT, K, m0, n0, 0, tid);
  wait_async0();
  __syncthreads();
  for (int k0 = 0; k0 < K; k0 += 64) {
    int cur = (k0 >> 6) & 1;
    if (k0 + 64 < K) gemm_stage(smem, cur ^ 1, A, WT, K, m0, n0, k0 + 64, tid);
    gemm_compute(smem, cur, wr, wc, half, l15, acc);
    wait_async0();
    __syncthreads();
  }

  // scatter into q/k/v head layouts; fold 1/sqrt(hd) into Q
#pragma unroll
  for (int j = 0; j < 2; j++) {
    int nG = n0 + wc * 32 + j * 16 + l15;
    float bv = bias[nG];
    int part = nG >> 10;                 // 0=q 1=k 2=v
    int c = nG & 1023;
    int h = c >> 6, d = c & 63;
#pragma unroll
    for (int i = 0; i < 2; i++)
#pragma unroll
      for (int r = 0; r < 8; r++) {
        int row = m0 + wr * 32 + i * 16 + r + 8 * half;
        int b = row >> 11, t = row & 2047;
        float v = acc[i][j][r] + bv;
        int bh = b * N_HEAD + h;
        if (part == 0)
          Qo[((size_t)bh * SEQ + t) * HEAD_DIM + d] = (bf16)(v * 0.125f);
        else if (part == 1)
          Ko[((size_t)bh * SEQ + t) * HEAD_DIM + d] = (bf16)v;
        else
          Vt[((size_t)bh * HEAD_DIM + d) * SEQ + t] = (bf16)v;  // transposed
      }
  }
}

// ---------------------------------------------------------------- attention
// LDS arena (bytes): Kbuf[2] @ 0 / 5120 ([key][80]), Vbuf[2] @ 10240 / 15360
// ([dim][40]), P @ 20480 ([wave*16+row][40]). Total 30 KiB.
#define ATTN_LDS_BYTES 30720
#define K_BUF(b)  ((b) * 5120)
#define V_BUF(b)  (10240 + (b) * 5120)
#define P_BASE    20480

__global__ void __launch_bounds__(256)
attn_fwd(const bf16* __restrict__ Q, const bf16* __restrict__ Kt,
         const bf16* __restrict__ Vt, bf16* __restrict__ Oa) {
  extern __shared__ char smem[];
  const int tid  = threadIdx.x;
  const int lane = tid & 31, wave = tid >> 5;
  const int half = lane >> 4, l15 = lane & 15;
  const int bh   = blockIdx.x;                  // b*16 + h
  const int qblk = blockIdx.y;
  const int qBase = qblk * 128 + wave * 16;

  // Q fragments (A layout), dims 0..31 / 32..63; softmax scale pre-folded
  const bf16* qrow = Q + ((size_t)bh * SEQ + qBase + l15) * HEAD_DIM;
  v16bf qf0 = make16(*(const bf16x8*)(qrow + half * 16),
                     *(const bf16x8*)(qrow + half * 16 + 8));
  v16bf qf1 = make16(*(const bf16x8*)(qrow + 32 + half * 16),
                     *(const bf16x8*)(qrow + 32 + half * 16 + 8));

  v8f o[4] = {{}, {}, {}, {}};
  float mrow[8], lrow[8];
#pragma unroll
  for (int r = 0; r < 8; r++) { mrow[r] = -__builtin_inff(); lrow[r] = 0.f; }

  const int kr = tid >> 3, kc8 = (tid & 7) * 8;   // K stage: 32x64, 8 elems/thr
  const int vr = tid >> 2, vc8 = (tid & 3) * 8;   // V stage: 64x32, 8 elems/thr

  auto stageKV = [&](int buf, int kt) {
    async_b128(K_BUF(buf) + (unsigned)(kr * 80 + kc8) * 2,
               Kt + ((size_t)bh * SEQ + kt * 32 + kr) * HEAD_DIM + kc8);
    async_b128(V_BUF(buf) + (unsigned)(vr * 40 + vc8) * 2,
               Vt + ((size_t)bh * HEAD_DIM + vr) * SEQ + kt * 32 + vc8);
  };

  const int nkt = qblk * 4 + 4;                   // causal key-tile count
  stageKV(0, 0);
  wait_async0();
  __syncthreads();
  for (int kt = 0; kt < nkt; kt++) {
    int cur = kt & 1;
    if (kt + 1 < nkt) stageKV(cur ^ 1, kt + 1);

    if (kt * 32 <= qBase + 15) {                  // wave-uniform causal skip
      const char* Ksb = smem + K_BUF(cur);
      const char* Vsb = smem + V_BUF(cur);
      char* Psb = smem + P_BASE;

      // ---- S = Q*K^T : two key groups x two hd chunks
      v8f s0 = {}, s1 = {};
#pragma unroll
      for (int g = 0; g < 2; g++) {
        int key = g * 16 + l15;
        v16bf kf0 = make16(*(const bf16x8*)(Ksb + (key * 80 + half * 16) * 2),
                           *(const bf16x8*)(Ksb + (key * 80 + half * 16 + 8) * 2));
        v16bf kf1 = make16(*(const bf16x8*)(Ksb + (key * 80 + 32 + half * 16) * 2),
                           *(const bf16x8*)(Ksb + (key * 80 + 32 + half * 16 + 8) * 2));
        v8f s = (g == 0) ? s0 : s1;
        s = wmma_bf16(qf0, kf0, s);
        s = wmma_bf16(qf1, kf1, s);
        if (g == 0) s0 = s; else s1 = s;
      }

      // ---- online softmax
#pragma unroll
      for (int r = 0; r < 8; r++) {
        int row  = qBase + r + 8 * half;
        int key0 = kt * 32 + l15;
        float v0 = s0[r]; if (key0 > row)      v0 = -__builtin_inff();
        float v1 = s1[r]; if (key0 + 16 > row) v1 = -__builtin_inff();
        float tmax = fmaxf(v0, v1);
        for (int off = 1; off < 16; off <<= 1)
          tmax = fmaxf(tmax, __shfl_xor(tmax, off));
        float mnew = fmaxf(mrow[r], tmax);
        float fac  = __expf(mrow[r] - mnew);
        mrow[r] = mnew;
        float p0 = __expf(v0 - mnew), p1 = __expf(v1 - mnew);
        float rs = p0 + p1;
        for (int off = 1; off < 16; off <<= 1) rs += __shfl_xor(rs, off);
        lrow[r] = lrow[r] * fac + rs;
#pragma unroll
        for (int t = 0; t < 4; t++) o[t][r] *= fac;
        int prow = wave * 16 + r + 8 * half;
        *(bf16*)(Psb + (prow * 40 + l15) * 2)      = (bf16)p0;
        *(bf16*)(Psb + (prow * 40 + 16 + l15) * 2) = (bf16)p1;
      }

      // ---- O += P*V : P re-read in A layout (wave-local), 4 dim tiles
      v16bf pf = make16(
          *(const bf16x8*)(Psb + ((wave * 16 + l15) * 40 + half * 16) * 2),
          *(const bf16x8*)(Psb + ((wave * 16 + l15) * 40 + half * 16 + 8) * 2));
#pragma unroll
      for (int t = 0; t < 4; t++) {
        int dim = t * 16 + l15;
        v16bf vf = make16(*(const bf16x8*)(Vsb + (dim * 40 + half * 16) * 2),
                          *(const bf16x8*)(Vsb + (dim * 40 + half * 16 + 8) * 2));
        o[t] = wmma_bf16(pf, vf, o[t]);
      }
    }
    wait_async0();
    __syncthreads();
  }

  // ---- normalize, store to [b, t, h*64 + d] bf16
  int b = bh >> 4, h = bh & 15;
#pragma unroll
  for (int r = 0; r < 8; r++) {
    int row = qBase + r + 8 * half;
    float inv = 1.f / lrow[r];
    size_t base = ((size_t)b * SEQ + row) * D_MODEL + h * HEAD_DIM;
#pragma unroll
    for (int t = 0; t < 4; t++)
      Oa[base + t * 16 + l15] = (bf16)(o[t][r] * inv);
  }
}

// ---------------------------------------------------------------- proj GEMM
__global__ void __launch_bounds__(256)
gemm_proj(const bf16* __restrict__ A, const bf16* __restrict__ WT,
          const float* __restrict__ bias, float* __restrict__ Out) {
  extern __shared__ char smem[];
  const int K = D_MODEL, N = D_MODEL;
  const int tid  = threadIdx.x;
  const int lane = tid & 31, wave = tid >> 5;
  const int half = lane >> 4, l15 = lane & 15;
  const int wr = wave >> 2, wc = wave & 3;
  const int m0 = blockIdx.x * 64, n0 = blockIdx.y * 128;

  v8f acc[2][2] = {{{}, {}}, {{}, {}}};

  gemm_stage(smem, 0, A, WT, K, m0, n0, 0, tid);
  wait_async0();
  __syncthreads();
  for (int k0 = 0; k0 < K; k0 += 64) {
    int cur = (k0 >> 6) & 1;
    if (k0 + 64 < K) gemm_stage(smem, cur ^ 1, A, WT, K, m0, n0, k0 + 64, tid);
    gemm_compute(smem, cur, wr, wc, half, l15, acc);
    wait_async0();
    __syncthreads();
  }

#pragma unroll
  for (int j = 0; j < 2; j++) {
    int nG = n0 + wc * 32 + j * 16 + l15;
    float bv = bias[nG];
#pragma unroll
    for (int i = 0; i < 2; i++)
#pragma unroll
      for (int r = 0; r < 8; r++) {
        int row = m0 + wr * 32 + i * 16 + r + 8 * half;
        Out[(size_t)row * N + nG] = acc[i][j][r] + bv;
      }
  }
}

// ---------------------------------------------------------------- launch
extern "C" void kernel_launch(void* const* d_in, const int* in_sizes, int n_in,
                              void* d_out, int out_size, void* d_ws, size_t ws_size,
                              hipStream_t stream) {
  const float* x     = (const float*)d_in[0];
  const float* Wqkv  = (const float*)d_in[1];
  const float* bqkv  = (const float*)d_in[2];
  const float* Wproj = (const float*)d_in[3];
  const float* bproj = (const float*)d_in[4];
  float* out = (float*)d_out;

  char* ws = (char*)d_ws;
  const size_t SZ_X    = (size_t)NROWS * D_MODEL * 2;          // 16 MiB
  const size_t SZ_WQKV = (size_t)D_MODEL * 3 * D_MODEL * 2;    //  6 MiB
  const size_t SZ_WPRJ = (size_t)D_MODEL * D_MODEL * 2;        //  2 MiB
  const size_t SZ_HEAD = (size_t)NROWS * D_MODEL * 2;          // 16 MiB each
  bf16* xb      = (bf16*)(ws);
  bf16* wqkvT   = (bf16*)(ws + SZ_X);
  bf16* wprojT  = (bf16*)(ws + SZ_X + SZ_WQKV);
  bf16* Qb      = (bf16*)(ws + SZ_X + SZ_WQKV + SZ_WPRJ);
  bf16* Kb      = (bf16*)(ws + SZ_X + SZ_WQKV + SZ_WPRJ + SZ_HEAD);
  bf16* VtB     = (bf16*)(ws + SZ_X + SZ_WQKV + SZ_WPRJ + 2 * SZ_HEAD);
  bf16* attnb   = (bf16*)(ws + SZ_X + SZ_WQKV + SZ_WPRJ + 3 * SZ_HEAD);

  int n = NROWS * D_MODEL;
  cvt_f32_bf16<<<(n + 255) / 256, 256, 0, stream>>>(x, xb, n);
  cvt_transpose<<<dim3(D_MODEL / 32, (3 * D_MODEL) / 32), dim3(32, 8), 0, stream>>>(
      Wqkv, wqkvT, D_MODEL, 3 * D_MODEL);
  cvt_transpose<<<dim3(D_MODEL / 32, D_MODEL / 32), dim3(32, 8), 0, stream>>>(
      Wproj, wprojT, D_MODEL, D_MODEL);

  gemm_qkv<<<dim3(NROWS / 64, (3 * D_MODEL) / 128), 256, GEMM_LDS_BYTES, stream>>>(
      xb, wqkvT, bqkv, Qb, Kb, VtB);

  attn_fwd<<<dim3(BATCH * N_HEAD, SEQ / 128), 256, ATTN_LDS_BYTES, stream>>>(
      Qb, Kb, VtB, attnb);

  gemm_proj<<<dim3(NROWS / 64, D_MODEL / 128), 256, GEMM_LDS_BYTES, stream>>>(
      attnb, wprojT, bproj, out);
}
